// TransitionUp_61478161875082
// MI455X (gfx1250) — matrix operand order repaired and checked by reading, not compiled, not run on hardware
//
#include <hip/hip_runtime.h>
#include <hip/hip_bf16.h>

// ---------------------------------------------------------------------------
// Problem constants (from reference)
// ---------------------------------------------------------------------------
#define BB   16
#define N1   4096
#define N2   1024
#define C2   256
#define OUTC 256
#define INCH 4096
#define BN_EPS 1e-5f

typedef __attribute__((ext_vector_type(16))) __bf16 v16bf;
typedef __attribute__((ext_vector_type(8)))  float  v8f;
typedef unsigned short u16;

union FragBF {
    v16bf v;
    uint4 q[2];
};

// round-to-nearest-even f32 -> bf16 (as raw u16)
__device__ __forceinline__ u16 f2bf(float f) {
    unsigned u = __float_as_uint(f);
    return (u16)((u + 0x7FFFu + ((u >> 16) & 1u)) >> 16);
}

// ---------------------------------------------------------------------------
// Kernel 0: convert W1 (f32, [256][4096]) to bf16
// ---------------------------------------------------------------------------
__global__ void k_convert_w1(const float* __restrict__ W1, u16* __restrict__ W1bf) {
    int i = blockIdx.x * blockDim.x + threadIdx.x;   // 0 .. 1048575
    W1bf[i] = f2bf(W1[i]);
}

// ---------------------------------------------------------------------------
// Kernel 1: fused 3-NN + inverse-distance interpolation.
// One block = 256 query points of one batch. xyz2[b] staged in LDS (12 KB).
// Writes interp transposed: interpT[b][c][n]  (bf16)  -> WMMA B-matrix layout.
// ---------------------------------------------------------------------------
__global__ void k_interp(const float* __restrict__ x2,
                         const float* __restrict__ xyz1,
                         const float* __restrict__ xyz2,
                         u16* __restrict__ interpT) {
    __shared__ float sx[N2], sy[N2], sz[N2];
    const int b    = blockIdx.x >> 4;          // 16 tiles per batch
    const int tile = blockIdx.x & 15;
    const int tid  = threadIdx.x;
    const int n    = tile * 256 + tid;         // query point index

    for (int i = tid; i < N2; i += 256) {
        const float* q = xyz2 + ((size_t)b * N2 + i) * 3;
        sx[i] = q[0]; sy[i] = q[1]; sz[i] = q[2];
    }
    __syncthreads();

    const float* p = xyz1 + ((size_t)b * N1 + n) * 3;
    const float px = p[0], py = p[1], pz = p[2];

    float d0 = 3.0e38f, d1 = 3.0e38f, d2 = 3.0e38f;
    int   i0 = 0, i1 = 0, i2 = 0;
    for (int j = 0; j < N2; ++j) {
        const float dx = px - sx[j], dy = py - sy[j], dz = pz - sz[j];
        const float d  = dx * dx + dy * dy + dz * dz;
        if (d < d2) {
            if (d < d1) {
                if (d < d0) { d2 = d1; i2 = i1; d1 = d0; i1 = i0; d0 = d; i0 = j; }
                else        { d2 = d1; i2 = i1; d1 = d;  i1 = j; }
            } else          { d2 = d;  i2 = j; }
        }
    }

    const float r0 = 1.0f / (d0 + 1e-8f);
    const float r1 = 1.0f / (d1 + 1e-8f);
    const float r2 = 1.0f / (d2 + 1e-8f);
    const float rs = 1.0f / (r0 + r1 + r2);
    const float w0 = r0 * rs, w1 = r1 * rs, w2 = r2 * rs;

    const float* x2b  = x2 + (size_t)b * N2 * C2;
    const float* row0 = x2b + (size_t)i0 * C2;
    const float* row1 = x2b + (size_t)i1 * C2;
    const float* row2 = x2b + (size_t)i2 * C2;

    u16* outBase = interpT + (size_t)b * C2 * N1 + n;   // stride N1 per channel
    #pragma unroll 4
    for (int c = 0; c < C2; ++c) {
        const float v = w0 * row0[c] + w1 * row1[c] + w2 * row2[c];
        outBase[(size_t)c * N1] = f2bf(v);              // coalesced along n across lanes
    }
}

// ---------------------------------------------------------------------------
// Kernel 2: batched GEMM  y[b][o][c] = sum_n W1[o][n] * interp[b][n][c] + b1[o]
// via v_wmma_f32_16x16x32_bf16. One wave per 32x32 output tile (2x2 frags).
// ---------------------------------------------------------------------------
__global__ void k_gemm_wmma(const u16* __restrict__ W1bf,
                            const u16* __restrict__ interpT,
                            const float* __restrict__ b1,
                            float* __restrict__ yws) {
    const int lane   = threadIdx.x & 31;
    const int wid    = blockIdx.x * 8 + (threadIdx.x >> 5);  // 0 .. 1023
    const int b      = wid >> 6;                             // batch
    const int m0     = ((wid >> 3) & 7) * 32;                // output-row (o) tile
    const int c0     = (wid & 7) * 32;                       // output-col (c) tile

    const int lanelo = lane & 15;
    const int lanehi = lane >> 4;
    const int kbA    = lanehi ? 16 : 0;   // A-matrix K base for this lane half
    const int kbB    = lanehi ? 16 : 0;   // B-matrix K base for this lane half

    // per-lane row base pointers (W1 row-major over K; interpT channel-major over n)
    const u16* aRow[2];
    const u16* bRow[2];
    #pragma unroll
    for (int mt = 0; mt < 2; ++mt)
        aRow[mt] = W1bf + (size_t)(m0 + mt * 16 + lanelo) * INCH + kbA;
    #pragma unroll
    for (int nt = 0; nt < 2; ++nt)
        bRow[nt] = interpT + ((size_t)b * C2 + c0 + nt * 16 + lanelo) * N1 + kbB;

    v8f acc[2][2];
    acc[0][0] = {}; acc[0][1] = {}; acc[1][0] = {}; acc[1][1] = {};

    for (int k0 = 0; k0 < INCH; k0 += 32) {
        FragBF a[2], bm[2];
        #pragma unroll
        for (int mt = 0; mt < 2; ++mt) {
            const u16* pa = aRow[mt] + k0;
            a[mt].q[0] = *(const uint4*)(pa);        // K = kbA+0 .. kbA+7
            a[mt].q[1] = *(const uint4*)(pa + 8);    // K = kbA+8 .. kbA+15
            __builtin_prefetch(pa + 64, 0, 1);       // global_prefetch next K-tiles
        }
        #pragma unroll
        for (int nt = 0; nt < 2; ++nt) {
            const u16* pb = bRow[nt] + k0;
            bm[nt].q[0] = *(const uint4*)(pb);       // K = kbB+0 .. kbB+7
            bm[nt].q[1] = *(const uint4*)(pb + 8);   // K = kbB+8 .. kbB+15
            __builtin_prefetch(pb + 64, 0, 1);
        }
        acc[0][0] = __builtin_amdgcn_wmma_f32_16x16x32_bf16(
            false, a[0].v, false, bm[0].v, (short)0, acc[0][0], false, false);
        acc[0][1] = __builtin_amdgcn_wmma_f32_16x16x32_bf16(
            false, a[0].v, false, bm[1].v, (short)0, acc[0][1], false, false);
        acc[1][0] = __builtin_amdgcn_wmma_f32_16x16x32_bf16(
            false, a[1].v, false, bm[0].v, (short)0, acc[1][0], false, false);
        acc[1][1] = __builtin_amdgcn_wmma_f32_16x16x32_bf16(
            false, a[1].v, false, bm[1].v, (short)0, acc[1][1], false, false);
    }

    // D layout: VGPR v holds row M = tile_m + v + 8*lanehi, col N = tile_c + lanelo
    #pragma unroll
    for (int mt = 0; mt < 2; ++mt) {
        #pragma unroll
        for (int nt = 0; nt < 2; ++nt) {
            #pragma unroll
            for (int v = 0; v < 8; ++v) {
                const int o = m0 + mt * 16 + v + 8 * lanehi;
                const int c = c0 + nt * 16 + lanelo;
                yws[((size_t)b * OUTC + o) * C2 + c] = acc[mt][nt][v] + b1[o];
            }
        }
    }
}

// ---------------------------------------------------------------------------
// Kernel 3: BatchNorm batch statistics. Channel = last index of yws[b][o][c].
// One thread per channel, coalesced sweep over the 4096 (b,o) rows.
// ---------------------------------------------------------------------------
__global__ void k_bn_stats(const float* __restrict__ yws,
                           const float* __restrict__ gamma,
                           const float* __restrict__ beta,
                           float* __restrict__ scale,
                           float* __restrict__ shift) {
    const int c = threadIdx.x;                 // 0..255, single block
    float s = 0.0f, ss = 0.0f;
    for (int i = 0; i < BB * OUTC; ++i) {
        const float v = yws[(size_t)i * C2 + c];
        s += v; ss += v * v;
    }
    const float inv_n = 1.0f / (float)(BB * OUTC);
    const float mean  = s * inv_n;
    const float var   = ss * inv_n - mean * mean;
    const float rstd  = rsqrtf(var + BN_EPS);
    const float sc    = gamma[c] * rstd;
    scale[c] = sc;
    shift[c] = beta[c] - mean * sc;
}

// ---------------------------------------------------------------------------
// Kernel 4: normalize + ReLU.  out[b][o][c] layout == yws layout (flat 1M).
// ---------------------------------------------------------------------------
__global__ void k_bn_apply(const float* __restrict__ yws,
                           const float* __restrict__ scale,
                           const float* __restrict__ shift,
                           float* __restrict__ out) {
    const int i = blockIdx.x * blockDim.x + threadIdx.x;
    const int c = i & (C2 - 1);
    out[i] = fmaxf(yws[i] * scale[c] + shift[c], 0.0f);
}

// ---------------------------------------------------------------------------
extern "C" void kernel_launch(void* const* d_in, const int* in_sizes, int n_in,
                              void* d_out, int out_size, void* d_ws, size_t ws_size,
                              hipStream_t stream) {
    // setup_inputs order: x1, x2, xyz1, xyz2, W1, b1, gamma, beta (all f32)
    const float* x2    = (const float*)d_in[1];
    const float* xyz1  = (const float*)d_in[2];
    const float* xyz2  = (const float*)d_in[3];
    const float* W1    = (const float*)d_in[4];
    const float* b1    = (const float*)d_in[5];
    const float* gamma = (const float*)d_in[6];
    const float* beta  = (const float*)d_in[7];
    float* out = (float*)d_out;

    // workspace layout
    char* ws = (char*)d_ws;
    u16*   W1bf    = (u16*)(ws);                              //  2 MB
    u16*   interpT = (u16*)(ws + ((size_t)2  << 20));         // 32 MB
    float* yws     = (float*)(ws + ((size_t)34 << 20));       //  4 MB
    float* scale   = (float*)(ws + ((size_t)38 << 20));       //  1 KB
    float* shift   = scale + C2;

    // 0) W1 -> bf16 (1M elems)
    k_convert_w1<<<dim3((OUTC * INCH) / 256), dim3(256), 0, stream>>>(W1, W1bf);

    // 1) fused 3-NN + interpolation, transposed bf16 output
    k_interp<<<dim3(BB * (N1 / 256)), dim3(256), 0, stream>>>(x2, xyz1, xyz2, interpT);

    // 2) batched WMMA GEMM: 1024 waves = 128 blocks x 8 waves
    k_gemm_wmma<<<dim3(128), dim3(256), 0, stream>>>(W1bf, interpT, b1, yws);

    // 3) BN statistics (one block, one thread per channel)
    k_bn_stats<<<dim3(1), dim3(C2), 0, stream>>>(yws, gamma, beta, scale, shift);

    // 4) normalize + ReLU -> output [B][OUT_CH][C2] flat (1M elems)
    k_bn_apply<<<dim3((BB * OUTC * C2) / 256), dim3(256), 0, stream>>>(yws, scale, shift, out);
}
